// SAGE_41128606826860
// MI455X (gfx1250) — matrix operand order repaired and checked by reading, not compiled, not run on hardware
//
#include <hip/hip_runtime.h>
#include <hip/hip_bf16.h>

typedef _Float16 v16h __attribute__((ext_vector_type(16)));
typedef float    v8f  __attribute__((ext_vector_type(8)));

#define GN   50000
#define GE   600000
#define FHID 128
#define FOUT 64
#define BNEPS 1e-5f

// ---------------------------------------------------------------------------
// Degree: cnt[dst] += 1 per edge
// ---------------------------------------------------------------------------
__global__ void degree_kernel(const int* __restrict__ dst, float* __restrict__ cnt, int E) {
    int e = blockIdx.x * blockDim.x + threadIdx.x;
    if (e < E) unsafeAtomicAdd(&cnt[dst[e]], 1.0f);
}

// ---------------------------------------------------------------------------
// Scatter: one wave (32 lanes) per edge; each lane moves float4 (128 floats)
// agg[dst, :] += w_e * x[src, :]
// ---------------------------------------------------------------------------
__global__ void scatter_kernel(const float* __restrict__ x, const int* __restrict__ src,
                               const int* __restrict__ dst, const float* __restrict__ w,
                               float* __restrict__ agg, int E) {
    int gid  = blockIdx.x * blockDim.x + threadIdx.x;
    int e    = gid >> 5;
    if (e >= E) return;
    int lane = gid & 31;
    int s = src[e], d = dst[e];
    float ww = w[e];
    const float4 v = *(const float4*)(x + (size_t)s * FHID + lane * 4);
    float* o = agg + (size_t)d * FHID + lane * 4;
    unsafeAtomicAdd(o + 0, v.x * ww);
    unsafeAtomicAdd(o + 1, v.y * ww);
    unsafeAtomicAdd(o + 2, v.z * ww);
    unsafeAtomicAdd(o + 3, v.w * ww);
}

// ---------------------------------------------------------------------------
// Pre-pack weights W[K,O] (f32 row-major) into per-lane WMMA B fragments (f16).
// Fragment index = ((ct*ksteps + kstep)*32 + lane), 16 halfs each.
// B layout (32x16 f16, wave32): VGPR r, lanes 0-15 -> K=2r,2r+1 (r<4), K=2r+8,2r+9 (r>=4) @ N=lane;
//                               lanes 16-31 -> same K + 8 @ N=lane-16.
// ---------------------------------------------------------------------------
__global__ void prep_w_kernel(const float* __restrict__ W, _Float16* __restrict__ out,
                              int K, int O) {
    int tid    = blockIdx.x * blockDim.x + threadIdx.x;
    int ksteps = K >> 5;
    int total  = (O >> 4) * ksteps * 32;
    if (tid >= total) return;
    int lane  = tid & 31;
    int t     = tid >> 5;
    int kstep = t % ksteps;
    int ct    = t / ksteps;
    int l     = lane & 15;
    int khalf = (lane >> 4) << 3;          // 0 or 8
    int col   = ct * 16 + l;
    _Float16* o = out + (size_t)tid * 16;
#pragma unroll
    for (int r = 0; r < 8; ++r) {
        int k = kstep * 32 + khalf + 2 * r + ((r >> 2) << 3);
        o[2 * r]     = (_Float16)W[(size_t)k * O + col];
        o[2 * r + 1] = (_Float16)W[(size_t)(k + 1) * O + col];
    }
}

// ---------------------------------------------------------------------------
// Build an A fragment (16x32 f16 layout) for this lane from row-major f32 A.
// Lane<16: K-runs [kb..kb+7] and [kb+16..kb+23] with kb=kstep*32;
// Lane>=16: same +8.  Elements scaled by `scale` (fused mean division).
// ---------------------------------------------------------------------------
__device__ __forceinline__ v16h make_afrag(const float* __restrict__ rowp, int kbase, float scale) {
    float4 q0 = *(const float4*)(rowp + kbase);
    float4 q1 = *(const float4*)(rowp + kbase + 4);
    float4 q2 = *(const float4*)(rowp + kbase + 16);
    float4 q3 = *(const float4*)(rowp + kbase + 20);
    v16h a;
    a[0]  = (_Float16)(q0.x * scale); a[1]  = (_Float16)(q0.y * scale);
    a[2]  = (_Float16)(q0.z * scale); a[3]  = (_Float16)(q0.w * scale);
    a[4]  = (_Float16)(q1.x * scale); a[5]  = (_Float16)(q1.y * scale);
    a[6]  = (_Float16)(q1.z * scale); a[7]  = (_Float16)(q1.w * scale);
    a[8]  = (_Float16)(q2.x * scale); a[9]  = (_Float16)(q2.y * scale);
    a[10] = (_Float16)(q2.z * scale); a[11] = (_Float16)(q2.w * scale);
    a[12] = (_Float16)(q3.x * scale); a[13] = (_Float16)(q3.y * scale);
    a[14] = (_Float16)(q3.z * scale); a[15] = (_Float16)(q3.w * scale);
    return a;
}

// ---------------------------------------------------------------------------
// Fused SAGE GEMM: out = (a1/max(cnt,1)) @ W1 + a2 @ W2 + bias
// One wave per 16-row strip covering ALL column tiles: each A fragment is
// loaded once per k-step and reused against COL_TILES B fragments (weights
// are a tiny packed table that lives in WGP$/L2).  K = 128 -> 4 k-steps,
// 2*COL_TILES WMMAs per k-step, 8*COL_TILES WMMAs per wave.
// ---------------------------------------------------------------------------
template <int COL_TILES>
__global__ void __launch_bounds__(128) sage_gemm_kernel(
    const float* __restrict__ a1, const float* __restrict__ cnt,
    const float* __restrict__ a2,
    const _Float16* __restrict__ w1f, const _Float16* __restrict__ w2f,
    const float* __restrict__ bias, float* __restrict__ out,
    int row_tiles) {
    const int K = 128, ksteps = 4, O = COL_TILES * 16;
    int wid  = threadIdx.x >> 5;
    int lane = threadIdx.x & 31;
    int rt   = blockIdx.x * 4 + wid;
    if (rt >= row_tiles) return;

    int row  = rt * 16 + (lane & 15);
    int koff = (lane >> 4) << 3;           // 0 or 8
    const float* r1 = a1 + (size_t)row * K;
    const float* r2 = a2 + (size_t)row * K;
    float s1 = 1.0f / fmaxf(cnt[row], 1.0f);

    const v16h* b1 = (const v16h*)w1f + lane;   // + (ct*ksteps + ks)*32
    const v16h* b2 = (const v16h*)w2f + lane;

    v8f acc[COL_TILES];
#pragma unroll
    for (int ct = 0; ct < COL_TILES; ++ct)
        acc[ct] = (v8f){0.f, 0.f, 0.f, 0.f, 0.f, 0.f, 0.f, 0.f};

#pragma unroll
    for (int ks = 0; ks < ksteps; ++ks) {
        int kbase = ks * 32 + koff;
        v16h af1 = make_afrag(r1, kbase, s1);
        v16h af2 = make_afrag(r2, kbase, 1.0f);
#pragma unroll
        for (int ct = 0; ct < COL_TILES; ++ct) {
            v16h bf1 = b1[(ct * ksteps + ks) * 32];
            acc[ct] = __builtin_amdgcn_wmma_f32_16x16x32_f16(false, af1, false, bf1,
                                                             (short)0, acc[ct], false, false);
            v16h bf2 = b2[(ct * ksteps + ks) * 32];
            acc[ct] = __builtin_amdgcn_wmma_f32_16x16x32_f16(false, af2, false, bf2,
                                                             (short)0, acc[ct], false, false);
        }
    }

    // C layout: VGPR r -> M=r (lanes 0-15) / M=r+8 (lanes 16-31), N=lane&15
    int l     = lane & 15;
    int rbase = rt * 16 + ((lane >> 4) << 3);
#pragma unroll
    for (int ct = 0; ct < COL_TILES; ++ct) {
        int n    = ct * 16 + l;
        float bv = bias[n];
#pragma unroll
        for (int r = 0; r < 8; ++r)
            out[(size_t)(rbase + r) * O + n] = acc[ct][r] + bv;
    }
}

// ---------------------------------------------------------------------------
// BatchNorm stats: column sums + sums of squares into stats[0..127], [128..255]
// ---------------------------------------------------------------------------
__global__ void bn_stats_kernel(const float* __restrict__ h, float* __restrict__ stats, int nrows) {
    int col = threadIdx.x;                 // 128 threads
    float s = 0.f, s2 = 0.f;
    for (int r = blockIdx.x; r < nrows; r += gridDim.x) {
        float v = h[(size_t)r * FHID + col];
        s += v;
        s2 += v * v;
    }
    unsafeAtomicAdd(&stats[col], s);
    unsafeAtomicAdd(&stats[FHID + col], s2);
}

// ---------------------------------------------------------------------------
// BatchNorm apply + ReLU (in place)
// ---------------------------------------------------------------------------
__global__ void bn_apply_kernel(float* __restrict__ h, const float* __restrict__ stats,
                                const float* __restrict__ gamma, const float* __restrict__ beta,
                                int nrows) {
    size_t idx = (size_t)blockIdx.x * blockDim.x + threadIdx.x;
    size_t total = (size_t)nrows * FHID;
    if (idx >= total) return;
    int col   = (int)(idx & (FHID - 1));
    float inv = 1.0f / (float)nrows;
    float mu  = stats[col] * inv;
    float var = stats[FHID + col] * inv - mu * mu;
    float v   = (h[idx] - mu) * rsqrtf(var + BNEPS) * gamma[col] + beta[col];
    h[idx] = v > 0.f ? v : 0.f;
}

// ---------------------------------------------------------------------------
extern "C" void kernel_launch(void* const* d_in, const int* in_sizes, int n_in,
                              void* d_out, int out_size, void* d_ws, size_t ws_size,
                              hipStream_t stream) {
    const int N = GN, E = GE;

    const float* x   = (const float*)d_in[0];
    const int*   src = (const int*)d_in[1];
    const int*   dst = src + E;
    const float* ew  = (const float*)d_in[2];
    const float* Wl0 = (const float*)d_in[3];
    const float* Wr0 = (const float*)d_in[4];
    const float* b0  = (const float*)d_in[5];
    const float* Wl1 = (const float*)d_in[6];
    const float* Wr1 = (const float*)d_in[7];
    const float* b1  = (const float*)d_in[8];
    const float* Wl2 = (const float*)d_in[9];
    const float* Wr2 = (const float*)d_in[10];
    const float* b2  = (const float*)d_in[11];
    const float* g0  = (const float*)d_in[12];
    const float* be0 = (const float*)d_in[13];
    const float* g1  = (const float*)d_in[14];
    const float* be1 = (const float*)d_in[15];
    float* out = (float*)d_out;

    // ---- workspace carve-up (256B aligned) ----
    char* ws = (char*)d_ws;
    size_t off = 0;
    auto carve = [&](size_t bytes) -> char* {
        char* p = ws + off;
        off += (bytes + 255) & ~(size_t)255;
        return p;
    };
    float*    cnt   = (float*)carve((size_t)N * 4);
    float*    agg   = (float*)carve((size_t)N * FHID * 4);
    float*    h0    = (float*)carve((size_t)N * FHID * 4);
    float*    h1    = (float*)carve((size_t)N * FHID * 4);
    float*    stats = (float*)carve(2 * FHID * 4);
    _Float16* wl0f  = (_Float16*)carve(128 * 128 * 2);
    _Float16* wr0f  = (_Float16*)carve(128 * 128 * 2);
    _Float16* wl1f  = (_Float16*)carve(128 * 128 * 2);
    _Float16* wr1f  = (_Float16*)carve(128 * 128 * 2);
    _Float16* wl2f  = (_Float16*)carve(128 * 64 * 2);
    _Float16* wr2f  = (_Float16*)carve(128 * 64 * 2);
    (void)ws_size; (void)in_sizes; (void)n_in; (void)out_size;

    // ---- pack weights into WMMA B-fragment layout ----
    prep_w_kernel<<<4, 256, 0, stream>>>(Wl0, wl0f, 128, 128);
    prep_w_kernel<<<4, 256, 0, stream>>>(Wr0, wr0f, 128, 128);
    prep_w_kernel<<<4, 256, 0, stream>>>(Wl1, wl1f, 128, 128);
    prep_w_kernel<<<4, 256, 0, stream>>>(Wr1, wr1f, 128, 128);
    prep_w_kernel<<<2, 256, 0, stream>>>(Wl2, wl2f, 128, 64);
    prep_w_kernel<<<2, 256, 0, stream>>>(Wr2, wr2f, 128, 64);

    // ---- degrees (shared by all layers) ----
    hipMemsetAsync(cnt, 0, (size_t)N * 4, stream);
    degree_kernel<<<(E + 255) / 256, 256, 0, stream>>>(dst, cnt, E);

    const int row_tiles   = N / 16;         // 3125
    const int gemm_blocks = (row_tiles + 3) / 4;
    const int scat_blocks = (E * 32 + 255) / 256;

    // ---- layer 0 ----
    hipMemsetAsync(agg, 0, (size_t)N * FHID * 4, stream);
    scatter_kernel<<<scat_blocks, 256, 0, stream>>>(x, src, dst, ew, agg, E);
    sage_gemm_kernel<8><<<gemm_blocks, 128, 0, stream>>>(
        agg, cnt, x, wl0f, wr0f, b0, h0, row_tiles);
    hipMemsetAsync(stats, 0, 2 * FHID * 4, stream);
    bn_stats_kernel<<<256, 128, 0, stream>>>(h0, stats, N);
    bn_apply_kernel<<<((size_t)N * FHID + 255) / 256, 256, 0, stream>>>(h0, stats, g0, be0, N);

    // ---- layer 1 ----
    hipMemsetAsync(agg, 0, (size_t)N * FHID * 4, stream);
    scatter_kernel<<<scat_blocks, 256, 0, stream>>>(h0, src, dst, ew, agg, E);
    sage_gemm_kernel<8><<<gemm_blocks, 128, 0, stream>>>(
        agg, cnt, h0, wl1f, wr1f, b1, h1, row_tiles);
    hipMemsetAsync(stats, 0, 2 * FHID * 4, stream);
    bn_stats_kernel<<<256, 128, 0, stream>>>(h1, stats, N);
    bn_apply_kernel<<<((size_t)N * FHID + 255) / 256, 256, 0, stream>>>(h1, stats, g1, be1, N);

    // ---- layer 2 (output, O = 64) ----
    hipMemsetAsync(agg, 0, (size_t)N * FHID * 4, stream);
    scatter_kernel<<<scat_blocks, 256, 0, stream>>>(h1, src, dst, ew, agg, E);
    sage_gemm_kernel<4><<<gemm_blocks, 128, 0, stream>>>(
        agg, cnt, h1, wl2f, wr2f, b2, out, row_tiles);
}
